// CausalSelfAttention_34591666602774
// MI455X (gfx1250) — matrix operand order repaired and checked by reading, not compiled, not run on hardware
//
#include <hip/hip_runtime.h>
#include <hip/hip_bf16.h>

typedef __attribute__((ext_vector_type(16))) _Float16 v16h;
typedef __attribute__((ext_vector_type(8)))  _Float16 v8h;
typedef __attribute__((ext_vector_type(8)))  float    v8f;
typedef __attribute__((ext_vector_type(4))) unsigned int u32x4;
typedef __attribute__((ext_vector_type(8))) int          i32x8;
typedef __attribute__((ext_vector_type(4))) int          i32x4;

#define B_ 4
#define T_ 2048
#define C_ 1024
#define H_ 16
#define D_ 64

#if __has_builtin(__builtin_amdgcn_tensor_load_to_lds) && __has_builtin(__builtin_amdgcn_s_wait_tensorcnt)
#define USE_TDM 1
#else
#define USE_TDM 0
#endif

// ---- fragment helpers (layouts per CDNA5 ISA 7.12.2) ----
// A 16x32 f16: lane l holds row (l&15); halves [0..7]=K[kb..kb+8), [8..15]=K[16+kb..), kb=8*(l>>4)
static __device__ __forceinline__ v16h frag_a(const _Float16* rowbase, int lane) {
  const int kb = (lane >> 4) * 8;
  union { v16h v; v8h h[2]; } u;
  u.h[0] = *(const v8h*)(rowbase + kb);
  u.h[1] = *(const v8h*)(rowbase + kb + 16);
  return u.v;
}
// B 32x16 f16 (source stored row-per-column, K contiguous): lane l holds col (l&15); halves [0..7]=K[kb..kb+8), [8..15]=K[16+kb..), kb=8*(l>>4)
static __device__ __forceinline__ v16h frag_b(const _Float16* rowbase, int lane) {
  const int kb = (lane >> 4) * 8;
  union { v16h v; v8h h[2]; } u;
  u.h[0] = *(const v8h*)(rowbase + kb);
  u.h[1] = *(const v8h*)(rowbase + kb + 16);
  return u.v;
}
static __device__ __forceinline__ v8f wmma_f16(v16h a, v16h b, v8f c) {
  return __builtin_amdgcn_wmma_f32_16x16x32_f16(false, a, false, b, (short)0, c, false, false);
}
static __device__ __forceinline__ v8f zero8() {
  v8f z;
#pragma unroll
  for (int i = 0; i < 8; ++i) z[i] = 0.f;
  return z;
}

// ---- fp32 -> f16 convert ----
__global__ void cvt_f16_kernel(const float* __restrict__ in, _Float16* __restrict__ out, int n) {
  int i = blockIdx.x * blockDim.x + threadIdx.x;
  const int stride = gridDim.x * blockDim.x;
  for (; i < n; i += stride) out[i] = (_Float16)in[i];
}

// ---- weight transpose [K,N] f32 -> [N,K] f16 (C_ x C_) ----
__global__ void transpose_w_kernel(const float* __restrict__ in, _Float16* __restrict__ out) {
  const int i = blockIdx.x * blockDim.x + threadIdx.x;  // i = n*C_ + k
  const int n = i >> 10;
  const int k = i & 1023;
  out[i] = (_Float16)in[k * C_ + n];
}

// ---- GEMM: out[M,N] = A[M,K](f16) * Bt[N,K]^T(f16) + bias(f32) ----
// block = 256 threads = 8 waves; wave tile 64x32; block tile 256x64
__global__ __launch_bounds__(256) void gemm_wmma_kernel(
    const _Float16* __restrict__ A, const _Float16* __restrict__ Bt,
    const float* __restrict__ bias, _Float16* __restrict__ outh,
    float* __restrict__ outf, int M, int N, int K) {
  const int lane = threadIdx.x & 31;
  const int wave = threadIdx.x >> 5;
  const int m0 = blockIdx.x * 256 + (wave >> 1) * 64;
  const int n0 = blockIdx.y * 64 + (wave & 1) * 32;
  const int rlo = lane & 15;
  const int half = lane >> 4;

  v8f acc[4][2];
#pragma unroll
  for (int i = 0; i < 4; ++i)
#pragma unroll
    for (int j = 0; j < 2; ++j) acc[i][j] = zero8();

  const _Float16* arow[4];
  const _Float16* brow[2];
#pragma unroll
  for (int i = 0; i < 4; ++i) arow[i] = A + (size_t)(m0 + 16 * i + rlo) * K;
#pragma unroll
  for (int j = 0; j < 2; ++j) brow[j] = Bt + (size_t)(n0 + 16 * j + rlo) * K;

  for (int k = 0; k < K; k += 32) {
    if (k + 128 < K) {
      __builtin_prefetch(arow[0] + k + 128, 0, 1);
      __builtin_prefetch(brow[0] + k + 128, 0, 1);
    }
    v16h af[4], bf[2];
#pragma unroll
    for (int i = 0; i < 4; ++i) af[i] = frag_a(arow[i] + k, lane);
#pragma unroll
    for (int j = 0; j < 2; ++j) bf[j] = frag_b(brow[j] + k, lane);
#pragma unroll
    for (int i = 0; i < 4; ++i)
#pragma unroll
      for (int j = 0; j < 2; ++j) acc[i][j] = wmma_f16(af[i], bf[j], acc[i][j]);
  }

#pragma unroll
  for (int j = 0; j < 2; ++j) {
    const int col = n0 + 16 * j + rlo;
    const float bv = bias[col];
#pragma unroll
    for (int i = 0; i < 4; ++i) {
#pragma unroll
      for (int v = 0; v < 8; ++v) {
        const int row = m0 + 16 * i + v + 8 * half;
        const float val = acc[i][j][v] + bv;
        if (outh) outh[(size_t)row * N + col] = (_Float16)val;
        else      outf[(size_t)row * N + col] = val;
      }
    }
  }
}

// ---- causal flash attention: Q,K,V f16 in [B,T,C] ([b,t,h*64+d]) -> O f16 [B,T,C] ----
// block = 128 threads = 4 waves; each wave owns a 16-query tile; key blocks of 32.
// K tile staged in LDS once per block via the Tensor Data Mover (all 4 waves reuse it).
__global__ __launch_bounds__(128) void attn_wmma_kernel(
    const _Float16* __restrict__ Q, const _Float16* __restrict__ Kh,
    const _Float16* __restrict__ Vh, _Float16* __restrict__ O) {
  __shared__ __align__(16) _Float16 sK[32 * D_];     // K tile row-major: [k][d], d-stride 64
  __shared__ __align__(16) _Float16 sVt[D_ * 32];    // V^T tile: [d][k], k-stride 32
  __shared__ __align__(16) _Float16 sP[4][16 * 32];  // per-wave P relayout scratch

  const int lane = threadIdx.x & 31;
  const int wave = threadIdx.x >> 5;
  const int b = blockIdx.y / H_;
  const int h = blockIdx.y % H_;
  const int q0 = blockIdx.x * 64 + wave * 16;
  const int ncol = lane & 15;
  const int half = lane >> 4;

  const size_t headoff = (size_t)b * T_ * C_ + (size_t)h * D_;

  // Q fragments for d 0..31 and 32..63 (row-major rows -> A layout)
  const _Float16* qrow = Q + headoff + (size_t)(q0 + ncol) * C_;
  const v16h qf0 = frag_a(qrow, lane);
  const v16h qf1 = frag_a(qrow + 32, lane);

  v8f acc[4];
#pragma unroll
  for (int n = 0; n < 4; ++n) acc[n] = zero8();
  float mrun[8], lrun[8];
#pragma unroll
  for (int v = 0; v < 8; ++v) { mrun[v] = -1e30f; lrun[v] = 0.f; }

  const int kbmax = (blockIdx.x * 64 + 63) >> 5;  // last key block (inclusive) for this block
  for (int kb = 0; kb <= kbmax; ++kb) {
    const int k0 = kb * 32;

#if USE_TDM
    // ---- TDM: DMA the 32x64 K tile for this key block into sK (wave 0 issues once) ----
    if (wave == 0) {
      const uint32_t ldsK  = (uint32_t)(uintptr_t)(void*)sK;
      const uint64_t gaddr = (uint64_t)(uintptr_t)(Kh + headoff + (size_t)k0 * C_);
      u32x4 g0;
      g0[0] = 1u;                                                // count=1, user mode
      g0[1] = ldsK;                                              // lds_addr
      g0[2] = (uint32_t)(gaddr & 0xFFFFFFFFu);                   // global_addr[31:0]
      g0[3] = (uint32_t)((gaddr >> 32) & 0x01FFFFFFu) | (2u << 30);  // global_addr[56:32] | type=2
      i32x8 g1;
      g1[0] = (1 << 16);                  // data_size = 2 bytes; wg_mask=0 (not in cluster)
      g1[1] = (D_ & 0xFFFF) << 16;        // tensor_dim0 = 64 elements
      g1[2] = (T_ & 0xFFFF) << 16;        // tensor_dim1 = 2048 rows
      g1[3] = (D_ << 16);                 // tile_dim0 = 64
      g1[4] = 32;                         // tile_dim1 = 32 rows
      g1[5] = C_;                         // tensor_dim0_stride = 1024 elements (row stride)
      g1[6] = 0;
      g1[7] = 0;
      i32x4 g2; i32x4 g3; i32x8 g4;
#pragma unroll
      for (int e = 0; e < 4; ++e) { g2[e] = 0; g3[e] = 0; }
#pragma unroll
      for (int e = 0; e < 8; ++e) g4[e] = 0;
      // 6-arg form (clang-23 / therock-10.0 headers): (u32x4, i32x8, i32x4, i32x4, i32x8, cpol)
      __builtin_amdgcn_tensor_load_to_lds(g0, g1, g2, g3, g4, 0);
    }
#else
    // fallback: cooperative copy of K tile into sK
    for (int c = threadIdx.x; c < 256; c += 128) {
      const int r = c >> 3;
      const int dc = (c & 7) * 8;
      *(v8h*)(sK + r * D_ + dc) = *(const v8h*)(Kh + headoff + (size_t)(k0 + r) * C_ + dc);
    }
#endif

    // ---- cooperative transpose of V[k0..k0+32)[0..64) into sVt[d][k] (overlaps TDM) ----
    {
      const int r = threadIdx.x & 31;       // key row within block
      const int dseg = threadIdx.x >> 5;    // 16 d-columns per wave
      const _Float16* vrow = Vh + headoff + (size_t)(k0 + r) * C_ + dseg * 16;
      const v8h v0 = *(const v8h*)(vrow);
      const v8h v1 = *(const v8h*)(vrow + 8);
#pragma unroll
      for (int e = 0; e < 8; ++e) {
        sVt[(dseg * 16 + e) * 32 + r] = v0[e];
        sVt[(dseg * 16 + 8 + e) * 32 + r] = v1[e];
      }
    }
#if USE_TDM
    if (wave == 0) __builtin_amdgcn_s_wait_tensorcnt((short)0);
#endif
    __syncthreads();

    if (k0 <= q0 + 15) {  // tile not fully above the diagonal
      // S = Q * K^T : K rows (contiguous in d) now come from LDS, shared by all 4 waves
      v8f s[2];
#pragma unroll
      for (int j = 0; j < 2; ++j) {
        const _Float16* krow = sK + (16 * j + ncol) * D_;
        const v16h kf0 = frag_b(krow, lane);       // d 0..31
        const v16h kf1 = frag_b(krow + 32, lane);  // d 32..63
        v8f sj = zero8();
        sj = wmma_f16(qf0, kf0, sj);
        sj = wmma_f16(qf1, kf1, sj);
        s[j] = sj;
      }
      // scale + causal mask (C-layout: row = v + 8*half, col = ncol)
#pragma unroll
      for (int v = 0; v < 8; ++v) {
        const int qg = q0 + v + 8 * half;
#pragma unroll
        for (int j = 0; j < 2; ++j) {
          const int kg = k0 + 16 * j + ncol;
          const float val = s[j][v] * 0.125f;  // 1/sqrt(64)
          s[j][v] = (kg <= qg) ? val : -1e30f;
        }
      }
      // online softmax (row reductions across the 16-lane half via shfl_xor)
#pragma unroll
      for (int v = 0; v < 8; ++v) {
        float mv = fmaxf(s[0][v], s[1][v]);
#pragma unroll
        for (int off = 8; off > 0; off >>= 1) mv = fmaxf(mv, __shfl_xor(mv, off, 32));
        const float mnew = fmaxf(mrun[v], mv);
        const float corr = __expf(mrun[v] - mnew);
        const float p0 = __expf(s[0][v] - mnew);
        const float p1 = __expf(s[1][v] - mnew);
        s[0][v] = p0; s[1][v] = p1;
        float rs = p0 + p1;
#pragma unroll
        for (int off = 8; off > 0; off >>= 1) rs += __shfl_xor(rs, off, 32);
        lrun[v] = lrun[v] * corr + rs;
        mrun[v] = mnew;
#pragma unroll
        for (int n = 0; n < 4; ++n) acc[n][v] *= corr;
      }
      // relayout P (C-layout f32) -> A-fragment via per-wave LDS (DS ops in-order per wave)
      _Float16* P = sP[wave];
#pragma unroll
      for (int v = 0; v < 8; ++v) {
        const int pr = v + 8 * half;
        P[pr * 32 + ncol]      = (_Float16)s[0][v];
        P[pr * 32 + 16 + ncol] = (_Float16)s[1][v];
      }
      __builtin_amdgcn_wave_barrier();
      const v16h pf = frag_a(P + ncol * 32, lane);
      // O += P * V  (V^T rows in LDS are contiguous in k -> B-fragment)
#pragma unroll
      for (int n = 0; n < 4; ++n) {
        const v16h vf = frag_b(sVt + (16 * n + ncol) * 32, lane);
        acc[n] = wmma_f16(pf, vf, acc[n]);
      }
    }
    __syncthreads();
  }

  // normalize and store [B,T,C] f16
#pragma unroll
  for (int v = 0; v < 8; ++v) {
    const float inv = (lrun[v] > 0.f) ? (1.f / lrun[v]) : 0.f;
    const int t = q0 + v + 8 * half;
    _Float16* orow = O + (size_t)(b * T_ + t) * C_ + h * D_;
#pragma unroll
    for (int n = 0; n < 4; ++n) orow[16 * n + ncol] = (_Float16)(acc[n][v] * inv);
  }
}

extern "C" void kernel_launch(void* const* d_in, const int* in_sizes, int n_in,
                              void* d_out, int out_size, void* d_ws, size_t ws_size,
                              hipStream_t stream) {
  (void)in_sizes; (void)n_in; (void)out_size; (void)ws_size;
  const float* x  = (const float*)d_in[0];
  const float* wq = (const float*)d_in[1];
  const float* bq = (const float*)d_in[2];
  const float* wk = (const float*)d_in[3];
  const float* bk = (const float*)d_in[4];
  const float* wv = (const float*)d_in[5];
  const float* bv = (const float*)d_in[6];
  const float* wo = (const float*)d_in[7];
  const float* bo = (const float*)d_in[8];
  float* out = (float*)d_out;

  const size_t MT = (size_t)B_ * T_;  // 8192 rows
  _Float16* p = (_Float16*)d_ws;
  _Float16* xh  = p; p += MT * C_;
  _Float16* wqT = p; p += (size_t)C_ * C_;
  _Float16* wkT = p; p += (size_t)C_ * C_;
  _Float16* wvT = p; p += (size_t)C_ * C_;
  _Float16* woT = p; p += (size_t)C_ * C_;
  _Float16* qb  = p; p += MT * C_;
  _Float16* kbf = p; p += MT * C_;
  _Float16* vbf = p; p += MT * C_;
  _Float16* ao  = p; p += MT * C_;

  // 1) convert x to f16; transpose weights to [N,K] f16
  cvt_f16_kernel<<<2048, 256, 0, stream>>>(x, xh, (int)(MT * C_));
  transpose_w_kernel<<<(C_ * C_) / 256, 256, 0, stream>>>(wq, wqT);
  transpose_w_kernel<<<(C_ * C_) / 256, 256, 0, stream>>>(wk, wkT);
  transpose_w_kernel<<<(C_ * C_) / 256, 256, 0, stream>>>(wv, wvT);
  transpose_w_kernel<<<(C_ * C_) / 256, 256, 0, stream>>>(wo, woT);

  // 2) QKV projections (f16 out, [B,T,C] = [b,t,h*64+d])
  dim3 gg((unsigned)(MT / 256), (unsigned)(C_ / 64));
  gemm_wmma_kernel<<<gg, 256, 0, stream>>>(xh, wqT, bq, qb,  nullptr, (int)MT, C_, C_);
  gemm_wmma_kernel<<<gg, 256, 0, stream>>>(xh, wkT, bk, kbf, nullptr, (int)MT, C_, C_);
  gemm_wmma_kernel<<<gg, 256, 0, stream>>>(xh, wvT, bv, vbf, nullptr, (int)MT, C_, C_);

  // 3) causal flash attention (TDM-staged K tiles)
  attn_wmma_kernel<<<dim3(T_ / 64, B_ * H_), 128, 0, stream>>>(qb, kbf, vbf, ao);

  // 4) output projection (f32 out -> d_out)
  gemm_wmma_kernel<<<gg, 256, 0, stream>>>(ao, woT, bo, nullptr, out, (int)MT, C_, C_);
}
